// CrossAttention_39109972197422
// MI455X (gfx1250) — compile-verified, matrix operand order checked by reading, not run on hardware
//
#include <hip/hip_runtime.h>

// ---------------------------------------------------------------------------
// Problem constants (from the reference)
// ---------------------------------------------------------------------------
static constexpr int B_    = 4;
static constexpr int C_    = 512;
static constexpr int H_    = 64;
static constexpr int W_    = 64;
static constexpr int HW_   = H_ * W_;    // 4096 query positions
static constexpr int XY_   = HW_;        // 4096 key/value positions
static constexpr int CQK_  = 64;         // Q/K channel dim
static constexpr int X_    = 64;         // softmax groups (per-x, over y)
static constexpr int Y_    = 64;

typedef __attribute__((ext_vector_type(16))) __bf16 v16bf;
typedef __attribute__((ext_vector_type(8)))  __bf16 v8bf;
typedef __attribute__((ext_vector_type(8)))  float  v8f;
typedef __attribute__((ext_vector_type(4)))  float  v4f;

// Load a v16bf WMMA A/B operand for one lane. CDNA5 16-bit A/B layout:
// lane = (row/col = L&15, hi = L>>4); VGPR0-3 hold K in [kbase+hi*8, +8),
// VGPR4-7 hold K in [kbase+16+hi*8, +8). Two contiguous 16B loads.
__device__ __forceinline__ v16bf load_op16(const __bf16* rowbase, int off0) {
    union { v16bf v; v8bf h[2]; } u;
    u.h[0] = *reinterpret_cast<const v8bf*>(rowbase + off0);
    u.h[1] = *reinterpret_cast<const v8bf*>(rowbase + off0 + 16);
    return u.v;
}

__device__ __forceinline__ v8f wmma_bf16(v16bf a, v16bf b, v8f c) {
    return __builtin_amdgcn_wmma_f32_16x16x32_bf16(
        /*neg_a=*/false, a, /*neg_b=*/false, b,
        /*c_mod=*/(short)0, c, /*reuse_a=*/false, /*reuse_b=*/false);
}

// ---------------------------------------------------------------------------
// Kernel 1: fp32 -> bf16 elementwise convert (weights)
// ---------------------------------------------------------------------------
__global__ void xattn_cvt_bf16(const float* __restrict__ in, __bf16* __restrict__ out, int n) {
    int i = blockIdx.x * 256 + threadIdx.x;
    if (i < n) out[i] = (__bf16)in[i];
}

// ---------------------------------------------------------------------------
// Kernel 2: LDS-tiled transpose  [b][c][p] fp32 -> [b][p][c] bf16.
// 64x64 tile per block; coalesced 256B row reads, coalesced 128B bf16 writes,
// pad-65 LDS rows for conflict-free transposed reads.
// ---------------------------------------------------------------------------
__global__ __launch_bounds__(256)
void xattn_tpose_cvt(const float* __restrict__ in, __bf16* __restrict__ out) {
    __shared__ float tile[64][65];
    const int b  = blockIdx.z;
    const int c0 = blockIdx.x * 64;
    const int p0 = blockIdx.y * 64;
    const int tx = threadIdx.x & 63;   // fast index
    const int ty = threadIdx.x >> 6;   // 4 rows per pass

#pragma unroll
    for (int r = 0; r < 64; r += 4) {
        int c = r + ty;
        tile[c][tx] = in[((long)b * C_ + c0 + c) * HW_ + p0 + tx];
    }
    __syncthreads();
#pragma unroll
    for (int r = 0; r < 64; r += 4) {
        int p = r + ty;
        out[((long)b * HW_ + p0 + p) * C_ + c0 + tx] = (__bf16)tile[tx][p];
    }
}

// ---------------------------------------------------------------------------
// Kernel 3: generic bf16 WMMA GEMM.  D[m][n] = sum_k A[m][k] * Bm[n][k] + bias
// A rows (length K) and Bm rows (length K) are K-contiguous bf16.
// One wave computes one 16x16 tile.  biasMode: 0=none, 1=per-n, 2=per-m.
// ---------------------------------------------------------------------------
__global__ __launch_bounds__(32)
void xattn_gemm_bf16(const __bf16* __restrict__ A, const __bf16* __restrict__ Bm,
                     __bf16* __restrict__ D, const float* __restrict__ bias,
                     int K, int ldD, int biasMode,
                     long aStride, long bStride, long dStride) {
    const int b    = blockIdx.z;
    const int m0   = blockIdx.x * 16;
    const int n0   = blockIdx.y * 16;
    const int lane = threadIdx.x & 31;
    const int col  = lane & 15;
    const int hi   = lane >> 4;
    const int koff = hi * 8;

    const __bf16* Ar = A  + b * aStride + (long)(m0 + col) * K;
    const __bf16* Br = Bm + b * bStride + (long)(n0 + col) * K;

    v8f acc = {0.f, 0.f, 0.f, 0.f, 0.f, 0.f, 0.f, 0.f};
    for (int k0 = 0; k0 < K; k0 += 32) {
        v16bf av = load_op16(Ar, k0 + koff);
        v16bf bv = load_op16(Br, k0 + koff);
        acc = wmma_bf16(av, bv, acc);
    }

    float bn = (biasMode == 1) ? bias[n0 + col] : 0.f;
#pragma unroll
    for (int v = 0; v < 8; ++v) {
        int m = v + 8 * hi;
        float add = (biasMode == 2) ? bias[m0 + m] : bn;
        D[b * dStride + (long)(m0 + m) * ldD + n0 + col] = (__bf16)(acc[v] + add);
    }
}

// ---------------------------------------------------------------------------
// Kernel 4: fused cross-attention.
//   One workgroup (8 waves) per (b, 16-query tile).
//   Per x in [0,64): every wave computes the full 16x64 score tile (8 WMMAs),
//   softmaxes each row over y via shfl_xor reductions, stages attn through its
//   private LDS slab, and accumulates attn @ V for its 64-channel slice.
//   K/V tiles for x+1 are prefetched (global_prefetch_b8) while x computes.
// ---------------------------------------------------------------------------
__global__ __launch_bounds__(256)
void xattn_attend(const __bf16* __restrict__ Qb, const __bf16* __restrict__ Kb,
                  const __bf16* __restrict__ Vb, const float* __restrict__ rgb,
                  const float* __restrict__ gamma_p, float* __restrict__ out) {
    __shared__ float smem[8][16][68];   // per-wave attn tile, pad-68 rows (bank-friendly)

    const int b    = blockIdx.y;
    const int q0   = blockIdx.x * 16;
    const int wv   = threadIdx.x >> 5;
    const int lane = threadIdx.x & 31;
    const int col  = lane & 15;
    const int hi   = lane >> 4;
    const int koff = hi * 8;
    const float gamma = gamma_p[0];

    // Q operand (A-matrix, 16 q x 64 c) is x-invariant: load once.
    const __bf16* Qrow = Qb + ((long)b * HW_ + q0 + col) * CQK_;
    v16bf aQ0 = load_op16(Qrow, 0  + koff);
    v16bf aQ1 = load_op16(Qrow, 32 + koff);

    const v8f vz = {0.f, 0.f, 0.f, 0.f, 0.f, 0.f, 0.f, 0.f};
    v8f oacc[4] = {vz, vz, vz, vz};     // 16 q x 64 c output slice (c0 = wv*64)
    const int c0 = wv * 64;

    const __bf16* Kbatch = Kb + (long)b * XY_ * CQK_;        // rows: xy, len 64
    const __bf16* Vbatch = Vb + (long)b * C_ * XY_;          // rows: c,  len XY

    for (int x = 0; x < X_; ++x) {
        // ---- prefetch next x's K block (8KB) and this wave's V slice ----
        if (x + 1 < X_) {
            const char* kn = (const char*)(Kbatch + (long)(x + 1) * Y_ * CQK_);
            __builtin_prefetch(kn + lane * 256, 0, 0);       // 32 lanes x 256B = 8KB
            const char* vn0 = (const char*)(Vbatch + (long)(c0 + lane) * XY_ + (x + 1) * Y_);
            const char* vn1 = (const char*)(Vbatch + (long)(c0 + 32 + lane) * XY_ + (x + 1) * Y_);
            __builtin_prefetch(vn0, 0, 0);                   // 64 rows x 128B
            __builtin_prefetch(vn1, 0, 0);
        }

        // ---- scores S[16 q x 64 y] = Q (16x64) x K^T (64x64) ----
        v8f sacc[4] = {vz, vz, vz, vz};
        const __bf16* Kbase = Kbatch + (long)x * Y_ * CQK_;
#pragma unroll
        for (int yt = 0; yt < 4; ++yt) {
            const __bf16* Krow = Kbase + (long)(yt * 16 + col) * CQK_;
            sacc[yt] = wmma_bf16(aQ0, load_op16(Krow, 0  + koff), sacc[yt]);
            sacc[yt] = wmma_bf16(aQ1, load_op16(Krow, 32 + koff), sacc[yt]);
        }

        // ---- softmax over the 64 y values of each row; write attn to LDS ----
#pragma unroll
        for (int v = 0; v < 8; ++v) {
            float m = fmaxf(fmaxf(sacc[0][v], sacc[1][v]),
                            fmaxf(sacc[2][v], sacc[3][v]));
            m = fmaxf(m, __shfl_xor(m, 1, 16));
            m = fmaxf(m, __shfl_xor(m, 2, 16));
            m = fmaxf(m, __shfl_xor(m, 4, 16));
            m = fmaxf(m, __shfl_xor(m, 8, 16));
            float e0 = __expf(sacc[0][v] - m);
            float e1 = __expf(sacc[1][v] - m);
            float e2 = __expf(sacc[2][v] - m);
            float e3 = __expf(sacc[3][v] - m);
            float s = e0 + e1 + e2 + e3;
            s += __shfl_xor(s, 1, 16);
            s += __shfl_xor(s, 2, 16);
            s += __shfl_xor(s, 4, 16);
            s += __shfl_xor(s, 8, 16);
            float r = __builtin_amdgcn_rcpf(s);
            int row = v + 8 * hi;
            smem[wv][row][0 * 16 + col] = e0 * r;
            smem[wv][row][1 * 16 + col] = e1 * r;
            smem[wv][row][2 * 16 + col] = e2 * r;
            smem[wv][row][3 * 16 + col] = e3 * r;
        }
        // Same-wave LDS store->load: DS ops are in-order per wave, no barrier.

        // ---- PV: oacc += attn (16x64) x V (64 xy x 64 c slice) ----
#pragma unroll
        for (int k = 0; k < 2; ++k) {
            const float* rp = &smem[wv][col][0];
            int o0 = k * 32 + koff;
            v4f f0 = *reinterpret_cast<const v4f*>(rp + o0);
            v4f f1 = *reinterpret_cast<const v4f*>(rp + o0 + 4);
            v4f f2 = *reinterpret_cast<const v4f*>(rp + o0 + 16);
            v4f f3 = *reinterpret_cast<const v4f*>(rp + o0 + 20);
            v16bf aA;
#pragma unroll
            for (int i = 0; i < 4; ++i) {
                aA[i]      = (__bf16)f0[i];
                aA[4 + i]  = (__bf16)f1[i];
                aA[8 + i]  = (__bf16)f2[i];
                aA[12 + i] = (__bf16)f3[i];
            }
#pragma unroll
            for (int nt = 0; nt < 4; ++nt) {
                const __bf16* Vrow = Vbatch + (long)(c0 + nt * 16 + col) * XY_ + x * Y_;
                oacc[nt] = wmma_bf16(aA, load_op16(Vrow, k * 32 + koff), oacc[nt]);
            }
        }
    }

    // ---- epilogue: out = rgb + gamma * attended ----
#pragma unroll
    for (int nt = 0; nt < 4; ++nt) {
#pragma unroll
        for (int v = 0; v < 8; ++v) {
            int c  = c0 + nt * 16 + col;
            int hw = q0 + v + 8 * hi;
            long idx = ((long)b * C_ + c) * HW_ + hw;
            out[idx] = rgb[idx] + gamma * oacc[nt][v];
        }
    }
}

// ---------------------------------------------------------------------------
// Host launcher
// ---------------------------------------------------------------------------
extern "C" void kernel_launch(void* const* d_in, const int* in_sizes, int n_in,
                              void* d_out, int out_size, void* d_ws, size_t ws_size,
                              hipStream_t stream) {
    const float* rgb   = (const float*)d_in[0];
    const float* chm   = (const float*)d_in[1];
    const float* Wq    = (const float*)d_in[2];
    const float* bq    = (const float*)d_in[3];
    const float* Wk    = (const float*)d_in[4];
    const float* bk    = (const float*)d_in[5];
    const float* Wv    = (const float*)d_in[6];
    const float* bv    = (const float*)d_in[7];
    const float* gamma = (const float*)d_in[8];
    float* out = (float*)d_out;

    // Workspace carve-up (~52.6 MiB total, 256B-aligned slabs).
    char* ws = (char*)d_ws;
    size_t off = 0;
    auto alloc = [&](size_t bytes) -> void* {
        void* p = ws + off;
        off += (bytes + 255) & ~(size_t)255;
        return p;
    };
    __bf16* rgbT = (__bf16*)alloc((size_t)B_ * HW_ * C_ * 2);   // [b][p][c]
    __bf16* chmT = (__bf16*)alloc((size_t)B_ * XY_ * C_ * 2);   // [b][p][c]
    __bf16* WqB  = (__bf16*)alloc((size_t)CQK_ * C_ * 2);
    __bf16* WkB  = (__bf16*)alloc((size_t)CQK_ * C_ * 2);
    __bf16* WvB  = (__bf16*)alloc((size_t)C_ * C_ * 2);
    __bf16* Qb   = (__bf16*)alloc((size_t)B_ * HW_ * CQK_ * 2); // [b][q][cqk]
    __bf16* Kb   = (__bf16*)alloc((size_t)B_ * XY_ * CQK_ * 2); // [b][xy][cqk]
    __bf16* Vb   = (__bf16*)alloc((size_t)B_ * C_ * XY_ * 2);   // [b][c][xy]

    // 1) weights -> bf16
    {
        int nq = CQK_ * C_, nv = C_ * C_;
        xattn_cvt_bf16<<<(nq + 255) / 256, 256, 0, stream>>>(Wq, WqB, nq);
        xattn_cvt_bf16<<<(nq + 255) / 256, 256, 0, stream>>>(Wk, WkB, nq);
        xattn_cvt_bf16<<<(nv + 255) / 256, 256, 0, stream>>>(Wv, WvB, nv);
    }
    // 2) features -> transposed bf16 (LDS-tiled, coalesced both sides)
    {
        dim3 tg(C_ / 64, HW_ / 64, B_);
        xattn_tpose_cvt<<<tg, 256, 0, stream>>>(rgb, rgbT);
        xattn_tpose_cvt<<<tg, 256, 0, stream>>>(chm, chmT);
    }
    // 3) projections (WMMA GEMMs)
    // Q[b][q][o]   = rgbT[b][q][:] . Wq[o][:]   (M=HW, N=64,  K=512, bias per-n)
    xattn_gemm_bf16<<<dim3(HW_ / 16, CQK_ / 16, B_), 32, 0, stream>>>(
        rgbT, WqB, Qb, bq, C_, CQK_, 1,
        (long)HW_ * C_, 0L, (long)HW_ * CQK_);
    // K[b][xy][o]  = chmT[b][xy][:] . Wk[o][:]
    xattn_gemm_bf16<<<dim3(XY_ / 16, CQK_ / 16, B_), 32, 0, stream>>>(
        chmT, WkB, Kb, bk, C_, CQK_, 1,
        (long)XY_ * C_, 0L, (long)XY_ * CQK_);
    // V[b][o][xy]  = Wv[o][:] . chmT[b][xy][:]  (M=512, N=XY, K=512, bias per-m)
    xattn_gemm_bf16<<<dim3(C_ / 16, XY_ / 16, B_), 32, 0, stream>>>(
        WvB, chmT, Vb, bv, C_, XY_, 2,
        0L, (long)XY_ * C_, (long)C_ * XY_);

    // 4) fused attention + residual
    xattn_attend<<<dim3(HW_ / 16, B_), 256, 0, stream>>>(
        Qb, Kb, Vb, rgb, gamma, out);

    (void)in_sizes; (void)n_in; (void)out_size; (void)ws_size;
}
